// GraphTransformerModel_79955111182813
// MI455X (gfx1250) — compile-verified, matrix operand order checked by reading, not run on hardware
//
#include <hip/hip_runtime.h>
#include <math.h>

// ---------------------------------------------------------------------------
// Types for WMMA
// ---------------------------------------------------------------------------
typedef __attribute__((ext_vector_type(16))) _Float16 v16h;
typedef __attribute__((ext_vector_type(8)))  _Float16 h8;
typedef __attribute__((ext_vector_type(8)))  float    v8f;

// Problem constants (from reference setup_inputs)
#define NN   50000
#define EE   800000
#define DD   128
#define HH   4
#define HID  32
#define HC   128    // H * HID
#define OUTC 40
#define OUTP 48     // padded to multiple of 16

// ---------------------------------------------------------------------------
// Helpers: monotonic float<->uint key for atomic max
// ---------------------------------------------------------------------------
__device__ __forceinline__ unsigned f2key(float f) {
  unsigned u = __float_as_uint(f);
  return (u & 0x80000000u) ? ~u : (u | 0x80000000u);
}
__device__ __forceinline__ float key2f(unsigned k) {
  unsigned u = (k & 0x80000000u) ? (k ^ 0x80000000u) : ~k;
  return __uint_as_float(u);
}
#define NEG_INF_KEY 0x007FFFFFu  // f2key(-inf)

// ---------------------------------------------------------------------------
// Small utility kernels
// ---------------------------------------------------------------------------
__global__ void init_u32_kernel(unsigned* __restrict__ p, unsigned val, int n) {
  int i = blockIdx.x * blockDim.x + threadIdx.x;
  if (i < n) p[i] = val;
}

__global__ void f32_to_f16_kernel(const float* __restrict__ in,
                                  _Float16* __restrict__ out, int n) {
  int i = blockIdx.x * blockDim.x + threadIdx.x;
  if (i < n) out[i] = (_Float16)in[i];
}

// W is K x M row-major fp32; produce WT = Mpad x K row-major f16 (B^T layout)
__global__ void transpose_f16_kernel(const float* __restrict__ W,
                                     _Float16* __restrict__ WT,
                                     int K, int M, int Mpad) {
  int idx = blockIdx.x * blockDim.x + threadIdx.x;
  if (idx >= Mpad * K) return;
  int m = idx / K;
  int k = idx % K;
  WT[idx] = (m < M) ? (_Float16)W[(size_t)k * M + m] : (_Float16)0.0f;
}

// ---------------------------------------------------------------------------
// WMMA GEMM: C[N x M] = A[N x K] (f16) @ B[K x M] (given as BT = M x K, f16)
//            + bias[M], fp32 accumulate & output.
// One wave computes one 16x16 tile. blockDim = (32, Mpad/16), grid = N/16.
// f16 A/B VGPR layout (ISA 7.12.2): lane l -> row/col = l%16,
// kbase = (l<16)?0:8, element j<4 -> K = kbase+2j{,+1}; j>=4 -> K = kbase+16+...
// i.e. two contiguous 8-half (16B) chunks at [kbase, kbase+8) and [kbase+16, +24).
// ---------------------------------------------------------------------------
__global__ __launch_bounds__(256)
void wmma_gemm_kernel(const _Float16* __restrict__ A,
                      const _Float16* __restrict__ BT,
                      const float* __restrict__ bias,
                      float* __restrict__ C,
                      int N, int K, int M) {
  const int lane = threadIdx.x;         // 0..31
  const int n0   = threadIdx.y * 16;    // column tile within block
  const int m0   = blockIdx.x * 16;     // row tile

  const int half  = lane >> 4;          // 0 or 1
  const int l16   = lane & 15;
  const int kbase = half * 8;

  int arow = m0 + l16; if (arow >= N) arow = N - 1;   // clamp (stores guarded)
  int bcol = n0 + l16; if (bcol >= M) bcol = M - 1;   // BT rows padded anyway

  const _Float16* ap = A  + (size_t)arow * K + kbase;
  const _Float16* bp = BT + (size_t)bcol * K + kbase;

  v8f acc = {};
  for (int kt = 0; kt < K; kt += 32) {
    h8 alo = *(const h8*)(ap + kt);
    h8 ahi = *(const h8*)(ap + kt + 16);
    h8 blo = *(const h8*)(bp + kt);
    h8 bhi = *(const h8*)(bp + kt + 16);
    v16h a, b;
#pragma unroll
    for (int j = 0; j < 8; ++j) {
      a[j] = alo[j]; a[j + 8] = ahi[j];
      b[j] = blo[j]; b[j + 8] = bhi[j];
    }
    acc = __builtin_amdgcn_wmma_f32_16x16x32_f16(
        /*neg_a=*/false, a, /*neg_b=*/false, b,
        /*c_mod=*/(short)0, acc, /*reuse_a=*/false, /*reuse_b=*/false);
  }

  // C/D layout: VGPR r -> row m0 + r + half*8, col n0 + l16
  const int col = n0 + l16;
  if (col < M) {
    const float bv = bias ? bias[col] : 0.0f;
#pragma unroll
    for (int r = 0; r < 8; ++r) {
      const int row = m0 + r + half * 8;
      if (row < N) C[(size_t)row * M + col] = acc[r] + bv;
    }
  }
}

// ---------------------------------------------------------------------------
// Edge pass 1: per-edge/head logits + scatter-max (monotonic uint atomicMax).
// One wave per edge; lane = h*8 + sub, sub covers 4 channels via float4.
// ---------------------------------------------------------------------------
__global__ __launch_bounds__(256)
void edge_logits_max_kernel(const float* __restrict__ q,
                            const float* __restrict__ k,
                            const int* __restrict__ srcI,
                            const int* __restrict__ dstI,
                            float* __restrict__ elog,
                            unsigned* __restrict__ mkey, int E) {
  const int lane = threadIdx.x & 31;
  const int e = blockIdx.x * 8 + (threadIdx.x >> 5);
  if (e >= E) return;
  const int s = srcI[e];
  const int d = dstI[e];
  const int h = lane >> 3;
  const int sub = lane & 7;

  const float4 q4 = *(const float4*)(q + (size_t)d * HC + h * HID + sub * 4);
  const float4 k4 = *(const float4*)(k + (size_t)s * HC + h * HID + sub * 4);
  float p = q4.x * k4.x + q4.y * k4.y + q4.z * k4.z + q4.w * k4.w;
  p += __shfl_xor(p, 1, 32);
  p += __shfl_xor(p, 2, 32);
  p += __shfl_xor(p, 4, 32);
  if (sub == 0) {
    const float logit = p * 0.17677669529663689f;  // 1/sqrt(32)
    elog[(size_t)e * HH + h] = logit;
    atomicMax(mkey + (size_t)d * HH + h, f2key(logit));
  }
}

// ---------------------------------------------------------------------------
// Edge pass 2: e = exp(logit - m[dst]) (stored in place) + scatter-sum denom.
// One thread per (edge, head).
// ---------------------------------------------------------------------------
__global__ __launch_bounds__(256)
void edge_exp_sum_kernel(const int* __restrict__ dstI,
                         float* __restrict__ elog,
                         const unsigned* __restrict__ mkey,
                         float* __restrict__ denom, int E) {
  const int t = blockIdx.x * blockDim.x + threadIdx.x;
  if (t >= E * HH) return;
  const int e = t >> 2;
  const int h = t & 3;
  const int d = dstI[e];
  const float m  = key2f(mkey[(size_t)d * HH + h]);
  const float ev = expf(elog[t] - m);
  elog[t] = ev;
  atomicAdd(denom + (size_t)d * HH + h, ev);
}

// ---------------------------------------------------------------------------
// Edge pass 3: acc[dst, c] += sum_h alpha[e,h] * v[src, h, c].
// One wave per edge; lane = channel c. Head-mean folded into finalize (1/H).
// ---------------------------------------------------------------------------
__global__ __launch_bounds__(256)
void edge_accum_kernel(const float* __restrict__ v,
                       const int* __restrict__ srcI,
                       const int* __restrict__ dstI,
                       const float* __restrict__ elog,
                       const float* __restrict__ denom,
                       float* __restrict__ acc, int E) {
  const int lane = threadIdx.x & 31;
  const int e = blockIdx.x * 8 + (threadIdx.x >> 5);
  if (e >= E) return;
  const int s = srcI[e];
  const int d = dstI[e];

  float ev = 0.0f, dn = 0.0f;
  if (lane < HH) {
    ev = elog[(size_t)e * HH + lane];
    dn = denom[(size_t)d * HH + lane];
  }
  const float a0 = __shfl(ev, 0, 32) / (__shfl(dn, 0, 32) + 1e-16f);
  const float a1 = __shfl(ev, 1, 32) / (__shfl(dn, 1, 32) + 1e-16f);
  const float a2 = __shfl(ev, 2, 32) / (__shfl(dn, 2, 32) + 1e-16f);
  const float a3 = __shfl(ev, 3, 32) / (__shfl(dn, 3, 32) + 1e-16f);

  const float* vp = v + (size_t)s * HC;
  const float out = a0 * vp[lane] + a1 * vp[HID + lane] +
                    a2 * vp[2 * HID + lane] + a3 * vp[3 * HID + lane];
  atomicAdd(acc + (size_t)d * HID + lane, out);
}

// ---------------------------------------------------------------------------
// Finalize: h = relu(acc/H + skip); also emit f16 copy for next WMMA GEMM.
// ---------------------------------------------------------------------------
__global__ __launch_bounds__(256)
void finalize_layer_kernel(const float* __restrict__ acc,
                           const float* __restrict__ skip,
                           float* __restrict__ hout,
                           _Float16* __restrict__ hh, int n) {
  const int i = blockIdx.x * blockDim.x + threadIdx.x;
  if (i >= n) return;
  float val = acc[i] * 0.25f + skip[i];
  val = fmaxf(val, 0.0f);
  hout[i] = val;
  hh[i] = (_Float16)val;
}

// ---------------------------------------------------------------------------
// Host-side orchestration
// ---------------------------------------------------------------------------
static inline size_t align256(size_t x) { return (x + 255) & ~(size_t)255; }

extern "C" void kernel_launch(void* const* d_in, const int* in_sizes, int n_in,
                              void* d_out, int out_size, void* d_ws, size_t ws_size,
                              hipStream_t stream) {
  (void)in_sizes; (void)n_in; (void)out_size; (void)ws_size;

  const float* x   = (const float*)d_in[0];
  const int*   ei  = (const int*)d_in[1];
  const float* wq1 = (const float*)d_in[2];  const float* bq1 = (const float*)d_in[3];
  const float* wk1 = (const float*)d_in[4];  const float* bk1 = (const float*)d_in[5];
  const float* wv1 = (const float*)d_in[6];  const float* bv1 = (const float*)d_in[7];
  const float* ws1 = (const float*)d_in[8];  const float* bs1 = (const float*)d_in[9];
  const float* wq2 = (const float*)d_in[10]; const float* bq2 = (const float*)d_in[11];
  const float* wk2 = (const float*)d_in[12]; const float* bk2 = (const float*)d_in[13];
  const float* wv2 = (const float*)d_in[14]; const float* bv2 = (const float*)d_in[15];
  const float* ws2 = (const float*)d_in[16]; const float* bs2 = (const float*)d_in[17];
  const float* wc  = (const float*)d_in[18]; const float* bc  = (const float*)d_in[19];

  const int* srcI = ei;
  const int* dstI = ei + EE;

  // Workspace carve-up
  char* w = (char*)d_ws;
  size_t off = 0;
  auto carve = [&](size_t bytes) { void* p = w + off; off += align256(bytes); return p; };

  _Float16* xh    = (_Float16*)carve((size_t)NN * DD * sizeof(_Float16));
  _Float16* wq1T  = (_Float16*)carve((size_t)HC  * DD  * sizeof(_Float16));
  _Float16* wk1T  = (_Float16*)carve((size_t)HC  * DD  * sizeof(_Float16));
  _Float16* wv1T  = (_Float16*)carve((size_t)HC  * DD  * sizeof(_Float16));
  _Float16* ws1T  = (_Float16*)carve((size_t)HID * DD  * sizeof(_Float16));
  _Float16* wq2T  = (_Float16*)carve((size_t)HC  * HID * sizeof(_Float16));
  _Float16* wk2T  = (_Float16*)carve((size_t)HC  * HID * sizeof(_Float16));
  _Float16* wv2T  = (_Float16*)carve((size_t)HC  * HID * sizeof(_Float16));
  _Float16* ws2T  = (_Float16*)carve((size_t)HID * HID * sizeof(_Float16));
  _Float16* wcT   = (_Float16*)carve((size_t)OUTP * HID * sizeof(_Float16));
  float*    qbuf  = (float*)carve((size_t)NN * HC * sizeof(float));
  float*    kbuf  = (float*)carve((size_t)NN * HC * sizeof(float));
  float*    vbuf  = (float*)carve((size_t)NN * HC * sizeof(float));
  float*    skip  = (float*)carve((size_t)NN * HID * sizeof(float));
  float*    elog  = (float*)carve((size_t)EE * HH * sizeof(float));
  unsigned* mkey  = (unsigned*)carve((size_t)NN * HH * sizeof(unsigned));
  float*    denom = (float*)carve((size_t)NN * HH * sizeof(float));
  float*    accb  = (float*)carve((size_t)NN * HID * sizeof(float));
  float*    hbuf  = (float*)carve((size_t)NN * HID * sizeof(float));
  _Float16* hh    = (_Float16*)carve((size_t)NN * HID * sizeof(_Float16));

  const int TB = 256;
  auto blocks = [](long long n, int tb) { return (unsigned)((n + tb - 1) / tb); };

  // --- one-time-per-call conversions ---
  f32_to_f16_kernel<<<blocks((long long)NN * DD, TB), TB, 0, stream>>>(x, xh, NN * DD);
  transpose_f16_kernel<<<blocks(HC  * DD,  TB), TB, 0, stream>>>(wq1, wq1T, DD,  HC,  HC);
  transpose_f16_kernel<<<blocks(HC  * DD,  TB), TB, 0, stream>>>(wk1, wk1T, DD,  HC,  HC);
  transpose_f16_kernel<<<blocks(HC  * DD,  TB), TB, 0, stream>>>(wv1, wv1T, DD,  HC,  HC);
  transpose_f16_kernel<<<blocks(HID * DD,  TB), TB, 0, stream>>>(ws1, ws1T, DD,  HID, HID);
  transpose_f16_kernel<<<blocks(HC  * HID, TB), TB, 0, stream>>>(wq2, wq2T, HID, HC,  HC);
  transpose_f16_kernel<<<blocks(HC  * HID, TB), TB, 0, stream>>>(wk2, wk2T, HID, HC,  HC);
  transpose_f16_kernel<<<blocks(HC  * HID, TB), TB, 0, stream>>>(wv2, wv2T, HID, HC,  HC);
  transpose_f16_kernel<<<blocks(HID * HID, TB), TB, 0, stream>>>(ws2, ws2T, HID, HID, HID);
  transpose_f16_kernel<<<blocks(OUTP * HID, TB), TB, 0, stream>>>(wc, wcT, HID, OUTC, OUTP);

  const dim3 gemmBlk128(32, HC / 16);   // 256 threads, 8 col tiles
  const dim3 gemmBlk32(32, HID / 16);   // 64 threads, 2 col tiles
  const dim3 gemmBlk48(32, OUTP / 16);  // 96 threads, 3 col tiles
  const unsigned gemmGrid = NN / 16;    // 3125 (exact)

  // ===================== Layer 1 =====================
  wmma_gemm_kernel<<<gemmGrid, gemmBlk128, 0, stream>>>(xh, wq1T, bq1, qbuf, NN, DD, HC);
  wmma_gemm_kernel<<<gemmGrid, gemmBlk128, 0, stream>>>(xh, wk1T, bk1, kbuf, NN, DD, HC);
  wmma_gemm_kernel<<<gemmGrid, gemmBlk128, 0, stream>>>(xh, wv1T, bv1, vbuf, NN, DD, HC);
  wmma_gemm_kernel<<<gemmGrid, gemmBlk32, 0, stream>>>(xh, ws1T, bs1, skip, NN, DD, HID);

  init_u32_kernel<<<blocks(NN * HH, TB), TB, 0, stream>>>(mkey, NEG_INF_KEY, NN * HH);
  init_u32_kernel<<<blocks(NN * HH, TB), TB, 0, stream>>>((unsigned*)denom, 0u, NN * HH);
  init_u32_kernel<<<blocks(NN * HID, TB), TB, 0, stream>>>((unsigned*)accb, 0u, NN * HID);

  edge_logits_max_kernel<<<EE / 8, TB, 0, stream>>>(qbuf, kbuf, srcI, dstI, elog, mkey, EE);
  edge_exp_sum_kernel<<<blocks((long long)EE * HH, TB), TB, 0, stream>>>(dstI, elog, mkey, denom, EE);
  edge_accum_kernel<<<EE / 8, TB, 0, stream>>>(vbuf, srcI, dstI, elog, denom, accb, EE);
  finalize_layer_kernel<<<blocks(NN * HID, TB), TB, 0, stream>>>(accb, skip, hbuf, hh, NN * HID);

  // ===================== Layer 2 =====================
  wmma_gemm_kernel<<<gemmGrid, gemmBlk128, 0, stream>>>(hh, wq2T, bq2, qbuf, NN, HID, HC);
  wmma_gemm_kernel<<<gemmGrid, gemmBlk128, 0, stream>>>(hh, wk2T, bk2, kbuf, NN, HID, HC);
  wmma_gemm_kernel<<<gemmGrid, gemmBlk128, 0, stream>>>(hh, wv2T, bv2, vbuf, NN, HID, HC);
  wmma_gemm_kernel<<<gemmGrid, gemmBlk32, 0, stream>>>(hh, ws2T, bs2, skip, NN, HID, HID);

  init_u32_kernel<<<blocks(NN * HH, TB), TB, 0, stream>>>(mkey, NEG_INF_KEY, NN * HH);
  init_u32_kernel<<<blocks(NN * HH, TB), TB, 0, stream>>>((unsigned*)denom, 0u, NN * HH);
  init_u32_kernel<<<blocks(NN * HID, TB), TB, 0, stream>>>((unsigned*)accb, 0u, NN * HID);

  edge_logits_max_kernel<<<EE / 8, TB, 0, stream>>>(qbuf, kbuf, srcI, dstI, elog, mkey, EE);
  edge_exp_sum_kernel<<<blocks((long long)EE * HH, TB), TB, 0, stream>>>(dstI, elog, mkey, denom, EE);
  edge_accum_kernel<<<EE / 8, TB, 0, stream>>>(vbuf, srcI, dstI, elog, denom, accb, EE);
  finalize_layer_kernel<<<blocks(NN * HID, TB), TB, 0, stream>>>(accb, skip, hbuf, hh, NN * HID);

  // ===================== Classifier =====================
  wmma_gemm_kernel<<<gemmGrid, gemmBlk48, 0, stream>>>(hh, wcT, bc, (float*)d_out, NN, HID, OUTC);
}
